// Attention_msa_visual_4664334483728
// MI455X (gfx1250) — matrix operand order repaired and checked by reading, not compiled, not run on hardware
//
#include <hip/hip_runtime.h>
#include <hip/hip_bf16.h>

typedef __bf16 bf16_t;
typedef __attribute__((ext_vector_type(16))) __bf16 v16bf;
typedef __attribute__((ext_vector_type(8)))  __bf16 v8bf;
typedef __attribute__((ext_vector_type(8)))  float   v8f;
typedef __attribute__((ext_vector_type(4)))  float   v4f;
typedef __attribute__((ext_vector_type(4)))  unsigned int u32x4;
typedef __attribute__((ext_vector_type(8)))  int          i32x8;
typedef __attribute__((ext_vector_type(4)))  int          i32x4;

#define NTOK 2048
#define CDIM 1024
#define HEADS 8
#define HD 128

#if __has_builtin(__builtin_amdgcn_tensor_load_to_lds)
#define HAS_TDM 1
#else
#define HAS_TDM 0
#endif

__device__ __forceinline__ v8f zero8() {
  v8f z;
#pragma unroll
  for (int i = 0; i < 8; ++i) z[i] = 0.f;
  return z;
}

__device__ __forceinline__ v8f wmma_bf16(v16bf a, v16bf b, v8f c) {
  // v_wmma_f32_16x16x32_bf16  (neg_a, A, neg_b, B, c_mod, C, reuse_a, reuse_b)
  return __builtin_amdgcn_wmma_f32_16x16x32_bf16(false, a, false, b, (short)0, c,
                                                 false, false);
}

// ---- CDNA5 async copy: GLOBAL_LOAD_ASYNC_TO_LDS_B128 (per-lane 16B, ASYNCcnt)
__device__ __forceinline__ void async_ld_b128(unsigned lds_addr, const void* g) {
  asm volatile("global_load_async_to_lds_b128 %0, %1, off"
               :: "v"(lds_addr), "v"(g)
               : "memory");
}

__device__ __forceinline__ void wait_async0() {
#if __has_builtin(__builtin_amdgcn_s_wait_asynccnt)
  __builtin_amdgcn_s_wait_asynccnt(0);
#else
  asm volatile("s_wait_asynccnt 0x0" ::: "memory");
#endif
}

__device__ __forceinline__ void wait_tensor0() {
#if __has_builtin(__builtin_amdgcn_s_wait_tensorcnt)
  __builtin_amdgcn_s_wait_tensorcnt(0);
#else
  asm volatile("s_wait_tensorcnt 0x0" ::: "memory");
#endif
}

__device__ __forceinline__ unsigned lds_addr_of(const void* p) {
  return (unsigned)(size_t)p;   // LDS aperture: low 32 bits are the LDS offset
}

#if HAS_TDM
// TDM 2D tile load: data_size = 8B units. Descriptor per CDNA5 ISA ch.8 (D#).
//  tile_d0 / tensor_d0 / stride0 in 8-byte units, tile_d1 / tensor_d1 in rows.
// This toolchain exposes the 6-arg builtin:
//   (uint32x4 g0, int32x8 g1, int32x4 g2, int32x4 g3, int32x8 pad, i32 cpol)
__device__ __forceinline__ void tdm_load_2d(unsigned lds, const void* gaddr,
                                            unsigned tile_d0, unsigned tile_d1,
                                            unsigned tdim0, unsigned tdim1,
                                            unsigned stride0) {
  unsigned long long ga = (unsigned long long)gaddr;
  u32x4 g0;
  g0[0] = 1u;                                   // count=1 (valid user D#)
  g0[1] = lds;                                  // lds_addr (bytes)
  g0[2] = (unsigned)(ga & 0xffffffffu);         // global_addr[31:0]
  g0[3] = (unsigned)((ga >> 32) & 0x1ffffffu)   // global_addr[56:32]
          | (2u << 30);                         // type = 2 ("image")
  i32x8 g1;
  g1[0] = (int)(3u << 16);                      // data_size = 3 (8 bytes)
  g1[1] = (int)((tdim0 & 0xffffu) << 16);       // tensor_dim0[15:0]
  g1[2] = (int)((tdim0 >> 16) | ((tdim1 & 0xffffu) << 16));  // td0[31:16] | td1[15:0]
  g1[3] = (int)((tdim1 >> 16) | (tile_d0 << 16));            // td1[31:16] | tile_dim0
  g1[4] = (int)(tile_d1 & 0xffffu);             // tile_dim1 (tile_dim2 = 0)
  g1[5] = (int)stride0;                         // tensor_dim0_stride[31:0]
  g1[6] = 0;                                    // stride0[47:32] | td1_stride[15:0]
  g1[7] = 0;
  i32x4 z4 = {0, 0, 0, 0};
  i32x8 z8;
#pragma unroll
  for (int i = 0; i < 8; ++i) z8[i] = 0;
  __builtin_amdgcn_tensor_load_to_lds(g0, g1, z4, z4, z8, 0);
}
#endif

// A operand as loaded here (16-bit, 16x32, k = 16*lh + s): elems 0..7 = K(16*lh..), elems 8..15 = K(16*lh+8..)
__device__ __forceinline__ v16bf ldA_bf(const bf16_t* p0, const bf16_t* p1) {
  v8bf a = *(const v8bf*)p0;
  v8bf b = *(const v8bf*)p1;
  v16bf r;
#pragma unroll
  for (int i = 0; i < 8; ++i) { r[i] = a[i]; r[8 + i] = b[i]; }
  return r;
}

__device__ __forceinline__ v16bf cvtA_f32(const float* p0, const float* p1) {
  const v4f* a = (const v4f*)p0;
  const v4f* b = (const v4f*)p1;
  v4f x0 = a[0], x1 = a[1], y0 = b[0], y1 = b[1];
  v16bf r;
#pragma unroll
  for (int i = 0; i < 4; ++i) {
    r[i] = (__bf16)x0[i]; r[4 + i] = (__bf16)x1[i];
    r[8 + i] = (__bf16)y0[i]; r[12 + i] = (__bf16)y1[i];
  }
  return r;
}

__device__ __forceinline__ v16bf cvtB_f32(const float* p) {
  const v4f* q = (const v4f*)p;
  v4f x0 = q[0], x1 = q[1], x2 = q[2], x3 = q[3];
  v16bf r;
#pragma unroll
  for (int i = 0; i < 4; ++i) {
    r[i] = (__bf16)x0[i]; r[4 + i] = (__bf16)x1[i];
    r[8 + i] = (__bf16)x2[i]; r[12 + i] = (__bf16)x3[i];
  }
  return r;
}

// ---------------------------------------------------------------- zero fill
__global__ void k_zero(float* __restrict__ p, size_t n) {
  size_t i = (size_t)blockIdx.x * blockDim.x + threadIdx.x;
  size_t st = (size_t)gridDim.x * blockDim.x;
  for (; i < n; i += st) p[i] = 0.f;
}

// ------------------------------------------------ v = x_cls @ W_v^T  (bf16 WMMA)
// block: 128 rows x 64 cols. W-chunk (64x32 f32) async-staged to LDS, shared by 8 waves.
__global__ void __launch_bounds__(256) k_gemm_v(
    const float* __restrict__ x, const float* __restrict__ W,
    float* __restrict__ out, bf16_t* __restrict__ vraw,
    float* __restrict__ normsq) {
  __shared__ __align__(32) float Wf[64][32];   // 8 KB
  const int tid  = threadIdx.x;
  const int lane = tid & 31;
  const int lm = lane & 15, lh = lane >> 4;
  const int w  = tid >> 5;
  const int jgrp = blockIdx.x >> 4;            // 16 col groups of 64
  const int mblk = blockIdx.x & 15;            // 16 row blocks of 128
  const int n0 = mblk * 128 + w * 16;
  const int j0 = jgrp * 64;
  const int h  = j0 >> 7;

  v8f acc[4];
#pragma unroll
  for (int t = 0; t < 4; ++t) acc[t] = zero8();

  for (int k0 = 0; k0 < CDIM; k0 += 32) {
    // stage W rows [2048+j0, +64) cols [k0, +32) : 512 x 16B, 2 per thread
#pragma unroll
    for (int i = 0; i < 2; ++i) {
      int idx = tid + 256 * i;
      int row = idx >> 3, c = idx & 7;
      async_ld_b128(lds_addr_of(&Wf[row][c * 4]),
                    W + (size_t)(2 * CDIM + j0 + row) * CDIM + k0 + c * 4);
    }
    wait_async0();
    __syncthreads();

    const float* pa = x + (size_t)(n0 + lm) * CDIM + k0 + 16 * lh;
    v16bf a = cvtA_f32(pa, pa + 8);
#pragma unroll
    for (int t = 0; t < 4; ++t) {
      v16bf b = cvtB_f32(&Wf[t * 16 + lm][16 * lh]);
      acc[t] = wmma_bf16(a, b, acc[t]);
    }
    __syncthreads();   // all reads done before next iteration's async overwrite
  }

#pragma unroll
  for (int r = 0; r < 8; ++r) {
    const int n = n0 + r + 8 * lh;
    float ss = 0.f;
#pragma unroll
    for (int t = 0; t < 4; ++t) {
      const int j = j0 + t * 16 + lm;
      float v = acc[t][r];
      out[(size_t)n * 2048 + 1024 + j] = v;                       // x_ori
      vraw[((size_t)h * NTOK + n) * HD + (j & 127)] = (bf16_t)v;
      ss += v * v;
    }
    atomicAdd(&normsq[h * NTOK + n], ss);
  }
}

// -------------------------------- normalize + build transposed bf16 copy
__global__ void k_norm(const bf16_t* __restrict__ vraw,
                       const float* __restrict__ normsq,
                       bf16_t* __restrict__ vbn, bf16_t* __restrict__ vT) {
  const size_t idx = (size_t)blockIdx.x * 256 + threadIdx.x;  // 8*2048*128
  if (idx >= (size_t)HEADS * NTOK * HD) return;
  const int h = (int)(idx >> 18);
  const int rem = (int)(idx & ((1 << 18) - 1));
  const int n = rem >> 7;
  const int d = rem & 127;
  bf16_t raw = vraw[idx];
  float inv = rsqrtf(normsq[h * NTOK + n]);
  vbn[idx] = (bf16_t)((float)raw * inv);
  vT[((size_t)(h * HD + d) << 11) + n] = raw;
}

// ------------- pass1: S, acr += S/8, L = sum exp(25S), x = (e @ v)/L
// B-tiles DMA'd to LDS by the Tensor Data Mover (wave 0), shared by 8 waves.
__global__ void __launch_bounds__(256) k_attn1(
    const bf16_t* __restrict__ vbn, const bf16_t* __restrict__ vT,
    float* __restrict__ acr, float* __restrict__ Lrow,
    float* __restrict__ out) {
  __shared__ __align__(32) bf16_t Bv[32][128];   // v-hat rows m0..m0+31     (8 KB)
  __shared__ __align__(32) bf16_t Bt[128][32];   // v-raw^T, d x m chunk     (8 KB)
  __shared__ __align__(32) bf16_t eS[8][16][32]; // exp staging, per wave    (8 KB)
  const int tid = threadIdx.x;
  const int lane = tid & 31, lm = lane & 15, lh = lane >> 4;
  const int w = tid >> 5;
  const int h = blockIdx.x >> 4;
  const int nb = blockIdx.x & 15;
  const int n0 = nb * 128 + w * 16;
  const bf16_t* vh  = vbn + ((size_t)h << 18);
  const bf16_t* vTh = vT  + ((size_t)h << 18);

  // hoist wave-private A tiles (16 rows x 128 K, loop-invariant)
  v16bf aS[4];
#pragma unroll
  for (int kk = 0; kk < 4; ++kk) {
    const bf16_t* pa = vh + (size_t)(n0 + lm) * HD + kk * 32 + 16 * lh;
    aS[kk] = ldA_bf(pa, pa + 8);
  }

  v8f xacc[8];
  float Lp[8];
#pragma unroll
  for (int t = 0; t < 8; ++t) { xacc[t] = zero8(); Lp[t] = 0.f; }

  for (int m0 = 0; m0 < NTOK; m0 += 32) {
#if HAS_TDM
    if (w == 0) {
      // Bv: 32 rows x 256B contiguous; Bt: 128 rows x 64B, 4KB row stride
      tdm_load_2d(lds_addr_of(&Bv[0][0]), vh + (size_t)m0 * HD,
                  /*tile_d0=*/32, /*tile_d1=*/32, /*td0=*/32, /*td1=*/NTOK,
                  /*stride0=*/32);
      tdm_load_2d(lds_addr_of(&Bt[0][0]), vTh + m0,
                  /*tile_d0=*/8, /*tile_d1=*/128, /*td0=*/512, /*td1=*/HEADS * HD,
                  /*stride0=*/512);
      wait_tensor0();
    }
#else
#pragma unroll
    for (int i = 0; i < 2; ++i) {
      int idx = tid + 256 * i;
      int row = idx >> 4, c = idx & 15;
      async_ld_b128(lds_addr_of(&Bv[row][c * 8]),
                    vh + (size_t)(m0 + row) * HD + c * 8);
      int d = idx >> 2, c2 = idx & 3;
      async_ld_b128(lds_addr_of(&Bt[d][c2 * 8]),
                    vTh + (size_t)d * 2048 + m0 + c2 * 8);
    }
    wait_async0();
#endif
    __syncthreads();

    v8f s0 = zero8(), s1 = zero8();
#pragma unroll
    for (int kk = 0; kk < 4; ++kk) {
      s0 = wmma_bf16(aS[kk], *(const v16bf*)&Bv[lm][kk * 32 + 16 * lh], s0);
      s1 = wmma_bf16(aS[kk], *(const v16bf*)&Bv[16 + lm][kk * 32 + 16 * lh], s1);
    }
    float e0[8], e1[8];
#pragma unroll
    for (int r = 0; r < 8; ++r) {
      const int n = n0 + r + 8 * lh;
      atomicAdd(&acr[(size_t)n * 2048 + m0 + lm],      s0[r] * 0.125f);
      atomicAdd(&acr[(size_t)n * 2048 + m0 + 16 + lm], s1[r] * 0.125f);
      e0[r] = __expf(25.f * s0[r]);
      e1[r] = __expf(25.f * s1[r]);
      Lp[r] += e0[r] + e1[r];
    }
#pragma unroll
    for (int r = 0; r < 8; ++r) {
      eS[w][r + 8 * lh][lm]      = (bf16_t)e0[r];
      eS[w][r + 8 * lh][16 + lm] = (bf16_t)e1[r];
    }
    __syncthreads();
    v16bf ea = ldA_bf(&eS[w][lm][16 * lh], &eS[w][lm][16 * lh + 8]);
#pragma unroll
    for (int dt = 0; dt < 8; ++dt) {
      xacc[dt] = wmma_bf16(ea, *(const v16bf*)&Bt[dt * 16 + lm][16 * lh], xacc[dt]);
    }
    __syncthreads();   // all LDS reads done before next iteration overwrites
  }

#pragma unroll
  for (int r = 0; r < 8; ++r) {
    float v = Lp[r];
    v += __shfl_xor(v, 1); v += __shfl_xor(v, 2);
    v += __shfl_xor(v, 4); v += __shfl_xor(v, 8);
    Lp[r] = v;
  }
  if (lm == 0) {
#pragma unroll
    for (int r = 0; r < 8; ++r) Lrow[h * NTOK + n0 + r + 8 * lh] = Lp[r];
  }
#pragma unroll
  for (int dt = 0; dt < 8; ++dt)
#pragma unroll
    for (int r = 0; r < 8; ++r) {
      const int n = n0 + r + 8 * lh;
      out[(size_t)n * 2048 + h * HD + dt * 16 + lm] = xacc[dt][r] / Lp[r];
    }
}

// -------------------- pass2: sim += exp(25S)/(8L)  (recompute S)
__global__ void __launch_bounds__(256) k_attn2(
    const bf16_t* __restrict__ vbn, const float* __restrict__ Lrow,
    float* __restrict__ simb) {
  __shared__ __align__(32) bf16_t Bv[32][128];   // 8 KB
  const int tid = threadIdx.x;
  const int lane = tid & 31, lm = lane & 15, lh = lane >> 4;
  const int w = tid >> 5;
  const int h = blockIdx.x >> 4;
  const int nb = blockIdx.x & 15;
  const int n0 = nb * 128 + w * 16;
  const bf16_t* vh = vbn + ((size_t)h << 18);

  v16bf aS[4];
#pragma unroll
  for (int kk = 0; kk < 4; ++kk) {
    const bf16_t* pa = vh + (size_t)(n0 + lm) * HD + kk * 32 + 16 * lh;
    aS[kk] = ldA_bf(pa, pa + 8);
  }

  float invL[8];
#pragma unroll
  for (int r = 0; r < 8; ++r)
    invL[r] = 0.125f / Lrow[h * NTOK + n0 + r + 8 * lh];

  for (int m0 = 0; m0 < NTOK; m0 += 32) {
#if HAS_TDM
    if (w == 0) {
      tdm_load_2d(lds_addr_of(&Bv[0][0]), vh + (size_t)m0 * HD,
                  /*tile_d0=*/32, /*tile_d1=*/32, /*td0=*/32, /*td1=*/NTOK,
                  /*stride0=*/32);
      wait_tensor0();
    }
#else
#pragma unroll
    for (int i = 0; i < 2; ++i) {
      int idx = tid + 256 * i;
      int row = idx >> 4, c = idx & 15;
      async_ld_b128(lds_addr_of(&Bv[row][c * 8]),
                    vh + (size_t)(m0 + row) * HD + c * 8);
    }
    wait_async0();
#endif
    __syncthreads();

    v8f s0 = zero8(), s1 = zero8();
#pragma unroll
    for (int kk = 0; kk < 4; ++kk) {
      s0 = wmma_bf16(aS[kk], *(const v16bf*)&Bv[lm][kk * 32 + 16 * lh], s0);
      s1 = wmma_bf16(aS[kk], *(const v16bf*)&Bv[16 + lm][kk * 32 + 16 * lh], s1);
    }
#pragma unroll
    for (int r = 0; r < 8; ++r) {
      const int n = n0 + r + 8 * lh;
      atomicAdd(&simb[(size_t)n * 2048 + m0 + lm],      __expf(25.f * s0[r]) * invL[r]);
      atomicAdd(&simb[(size_t)n * 2048 + m0 + 16 + lm], __expf(25.f * s1[r]) * invL[r]);
    }
    __syncthreads();
  }
}

// ----- finalize: out = mask*exp(sim) / sum(mask*exp(sim))   (softmax denom cancels)
__global__ void __launch_bounds__(256) k_final(
    const float* __restrict__ acr, const float* __restrict__ simb,
    float* __restrict__ out2) {
  __shared__ float buf[2048];
  __shared__ float red[256];
  const int n = blockIdx.x, tid = threadIdx.x;
  float part = 0.f;
  for (int m = tid; m < 2048; m += 256) {
    float e = __expf(simb[(size_t)n * 2048 + m]);
    float t = (acr[(size_t)n * 2048 + m] > 0.75f) ? e : 0.f;
    buf[m] = t;
    part += t;
  }
  red[tid] = part;
  __syncthreads();
  for (int s = 128; s > 0; s >>= 1) {
    if (tid < s) red[tid] += red[tid + s];
    __syncthreads();
  }
  const float inv = 1.f / red[0];
  for (int m = tid; m < 2048; m += 256)
    out2[(size_t)n * 2048 + m] = buf[m] * inv;
}

extern "C" void kernel_launch(void* const* d_in, const int* in_sizes, int n_in,
                              void* d_out, int out_size, void* d_ws, size_t ws_size,
                              hipStream_t stream) {
  (void)in_sizes; (void)n_in; (void)out_size; (void)ws_size;
  const float* x_cls = (const float*)d_in[0];
  const float* W_cls = (const float*)d_in[4];
  float* out = (float*)d_out;

  char* ws = (char*)d_ws;
  bf16_t* vraw  = (bf16_t*)(ws);                              // 4 MB
  bf16_t* vbn   = (bf16_t*)(ws + (size_t)(4  << 20));         // 4 MB
  bf16_t* vT    = (bf16_t*)(ws + (size_t)(8  << 20));         // 4 MB
  float*  normsq= (float*)(ws + (size_t)(12 << 20));          // 64 KB
  float*  Lrow  = (float*)(ws + (size_t)(12 << 20) + 65536);  // 64 KB
  float*  acr   = (float*)(ws + (size_t)(13 << 20));          // 16 MB
  float*  simb  = (float*)(ws + (size_t)(29 << 20));          // 16 MB

  // zero normsq + L + acr + sim  (bytes [12MB, 45MB))
  k_zero<<<2048, 256, 0, stream>>>((float*)(ws + (size_t)(12 << 20)),
                                   ((size_t)33 << 20) >> 2);
  k_gemm_v<<<256, 256, 0, stream>>>(x_cls, W_cls, out, vraw, normsq);
  k_norm<<<8192, 256, 0, stream>>>(vraw, normsq, vbn, vT);
  k_attn1<<<128, 256, 0, stream>>>(vbn, vT, acr, Lrow, out);
  k_attn2<<<128, 256, 0, stream>>>(vbn, Lrow, simb);
  k_final<<<2048, 256, 0, stream>>>(acr, simb, out + (size_t)2048 * 2048);
}
